// FPNSSD_77653008711788
// MI455X (gfx1250) — matrix operand order repaired and checked by reading, not compiled, not run on hardware
//
#include <hip/hip_runtime.h>
#include <hip/hip_bf16.h>
#include <math.h>

typedef __attribute__((ext_vector_type(16))) _Float16     v16h;
typedef __attribute__((ext_vector_type(8)))  float        v8f;
typedef __attribute__((ext_vector_type(4)))  unsigned int v4u;
typedef __attribute__((ext_vector_type(8)))  int          v8i;
typedef __attribute__((ext_vector_type(4)))  int          v4i;

#define NUM_CLASSES 21
#define TOPK 200
#define NPRIOR 8732

#ifndef __has_builtin
#define __has_builtin(x) 0
#endif
#if __has_builtin(__builtin_amdgcn_tensor_load_to_lds) && \
    __has_builtin(__builtin_amdgcn_s_wait_tensorcnt)
#define USE_TDM 1
#else
#define USE_TDM 0
#endif

// ---------------------------------------------------------------------------
// Implicit-GEMM convolution via WMMA f16 (f32 accumulate).
//   M = Cout, N = B*OH*OW, K = Cin*KH*KW
// One wave computes a 16(M) x 32(N) tile; the A (weight) fragment is fetched
// by the Tensor Data Mover into LDS once per K-step (reused by both WMMA
// issues / all 32 columns), while vector loads do the scattered im2col B path.
// Wave32 fragment layouts as used here (A and B share a contiguous k order):
//   A 16x32 f16 : lane row m = lane&15; half h -> K = (lane<16?0:16)+h
//   B 32x16 f16 : lane col n = lane&15; half h -> K = (lane<16?0:16)+h
//   C/D 16x16 f32: lane col n = lane&15; vgpr r -> row m = r + 8*(lane>>4)
// ---------------------------------------------------------------------------
__global__ __launch_bounds__(32)
void conv_wmma(const float* __restrict__ X, const float* __restrict__ Wt,
               const float* __restrict__ Bi, float* __restrict__ Y,
               int Bn, int Cin, int H, int Wd, int Cout, int OH, int OW,
               int KH, int KW, int stride, int pad, int dil, int relu)
{
#if USE_TDM
    __shared__ float lw[16 * 32];   // one 16(M) x 32(K) fp32 weight tile
#endif
    const int lane = threadIdx.x & 31;
    const int hi   = lane >> 4;          // lane half (0/1)
    const int lm   = lane & 15;
    const int NHW  = OH * OW;
    const int Ntot = Bn * NHW;
    const int KHW  = KH * KW;
    const int Ktot = Cin * KHW;
    const int n0   = blockIdx.x * 32;
    const int m0   = blockIdx.y * 16;

    // Decode the two output columns (pixels) this lane contributes to.
    int  cb[2], cy[2], cx[2];
    bool cok[2];
#pragma unroll
    for (int t = 0; t < 2; ++t) {
        int nc = n0 + t * 16 + lm;
        cok[t] = (nc < Ntot);
        int ncc = cok[t] ? nc : 0;
        cb[t] = ncc / NHW;
        int r = ncc - cb[t] * NHW;
        cy[t] = r / OW;
        cx[t] = r - cy[t] * OW;
    }
    const int iy00 = cy[0] * stride - pad, ix00 = cx[0] * stride - pad;
    const int iy01 = cy[1] * stride - pad, ix01 = cx[1] * stride - pad;
    const float* Xb0 = X + (size_t)cb[0] * Cin * H * Wd;
    const float* Xb1 = X + (size_t)cb[1] * Cin * H * Wd;

    const int  am   = m0 + lm;
    const bool amok = (am < Cout);
#if !USE_TDM
    const float* Wrow = Wt + (size_t)(amok ? am : 0) * Ktot;
#else
    const unsigned ldsoff = (unsigned)(unsigned long long)(uintptr_t)&lw[0];
#endif

    v8f acc0 = {}, acc1 = {};
    for (int k0 = 0; k0 < Ktot; k0 += 32) {
        // ---- A fragment ----
        v16h a;
#if USE_TDM
        {
            // WAR: previous step's ds reads must retire before the DMA rewrite.
            asm volatile("s_wait_dscnt 0x0" ::: "memory");
            unsigned long long ga =
                (unsigned long long)(uintptr_t)(Wt + (size_t)m0 * Ktot + k0);
            unsigned td0 = (unsigned)(Ktot - k0);   // K extent left (OOB -> 0)
            unsigned td1 = (unsigned)(Cout - m0);   // M extent left (OOB -> 0)
            v4u g0 = { 1u,                                   // count=1 descriptor
                       ldsoff,                               // lds_addr
                       (unsigned)ga,                         // global_addr[31:0]
                       (unsigned)(ga >> 32) | 0x80000000u }; // ga[56:32] | type=2
            v8i g1;
            g1[0] = (int)(2u << 16);                              // data_size=4B
            g1[1] = (int)((td0 & 0xFFFFu) << 16);                 // tensor_dim0 lo
            g1[2] = (int)(((td0 >> 16) & 0xFFFFu) |
                          ((td1 & 0xFFFFu) << 16));               // td0 hi | td1 lo
            g1[3] = (int)(((td1 >> 16) & 0xFFFFu) | (32u << 16)); // td1 hi | tile0=32
            g1[4] = 16;                                           // tile_dim1=16
            g1[5] = (int)(unsigned)Ktot;                          // dim0_stride lo
            g1[6] = 0;                                            // stride hi
            g1[7] = 0;
            v4i gz = { 0, 0, 0, 0 };
#if __clang_major__ >= 23
            v8i gz8 = { 0, 0, 0, 0, 0, 0, 0, 0 };
            __builtin_amdgcn_tensor_load_to_lds(g0, g1, gz, gz, gz8, 0);
#else
            __builtin_amdgcn_tensor_load_to_lds(g0, g1, gz, gz, 0);
#endif
            __builtin_amdgcn_s_wait_tensorcnt(0);
        }
#pragma unroll
        for (int h = 0; h < 16; ++h) {
            int kk = hi * 16 + h;
            a[h] = (_Float16)lw[lm * 32 + kk];
        }
#else
        if (k0 + 32 < Ktot) __builtin_prefetch(Wrow + k0 + 32, 0, 0);
#pragma unroll
        for (int h = 0; h < 16; ++h) {
            int k = k0 + hi * 16 + h;
            float av = (amok && k < Ktot) ? Wrow[k] : 0.0f;
            a[h] = (_Float16)av;
        }
#endif
        // ---- B fragments: implicit im2col, incremental (ci,ky,kx) carry ----
        const int kbase = k0 + hi * 16;
        int ci = kbase / KHW;
        int rr = kbase - ci * KHW;
        int ky = rr / KW;
        int kx = rr - ky * KW;
        v16h bfr0, bfr1;
#pragma unroll
        for (int h = 0; h < 16; ++h) {
            float v0 = 0.0f, v1 = 0.0f;
            if (kbase + h < Ktot) {
                const size_t coff = (size_t)ci * H;
                const int dy = ky * dil, dx = kx * dil;
                {
                    int iy = iy00 + dy, ix = ix00 + dx;
                    if (cok[0] && (unsigned)iy < (unsigned)H &&
                        (unsigned)ix < (unsigned)Wd)
                        v0 = Xb0[(coff + iy) * Wd + ix];
                }
                {
                    int iy = iy01 + dy, ix = ix01 + dx;
                    if (cok[1] && (unsigned)iy < (unsigned)H &&
                        (unsigned)ix < (unsigned)Wd)
                        v1 = Xb1[(coff + iy) * Wd + ix];
                }
            }
            bfr0[h] = (_Float16)v0;
            bfr1[h] = (_Float16)v1;
            if (++kx == KW) { kx = 0; if (++ky == KH) { ky = 0; ++ci; } }
        }
        acc0 = __builtin_amdgcn_wmma_f32_16x16x32_f16(false, a, false, bfr0,
                                                      (short)0, acc0, false, false);
        acc1 = __builtin_amdgcn_wmma_f32_16x16x32_f16(false, a, false, bfr1,
                                                      (short)0, acc1, false, false);
    }

#pragma unroll
    for (int t = 0; t < 2; ++t) {
        if (!cok[t]) continue;
        v8f ac = t ? acc1 : acc0;
        float* Yb = Y + (size_t)cb[t] * Cout * NHW + (size_t)cy[t] * OW + cx[t];
#pragma unroll
        for (int r = 0; r < 8; ++r) {
            int m = m0 + r + 8 * hi;
            if (m < Cout) {
                float v = ac[r] + Bi[m];
                if (relu) v = fmaxf(v, 0.0f);
                Yb[(size_t)m * NHW] = v;
            }
        }
    }
    (void)amok;
}

// ---------------------------------------------------------------------------
__global__ void maxpool2_kernel(const float* __restrict__ X, float* __restrict__ Y,
                                int BC, int H, int W, int OH, int OW)
{
    int i = blockIdx.x * blockDim.x + threadIdx.x;
    int tot = BC * OH * OW;
    if (i >= tot) return;
    int bc = i / (OH * OW);
    int r  = i - bc * OH * OW;
    int oy = r / OW, ox = r - (r / OW) * OW;
    const float* xp = X + (size_t)bc * H * W;
    float m = -INFINITY;
#pragma unroll
    for (int dy = 0; dy < 2; ++dy) {
        int iy = 2 * oy + dy;
        if (iy >= H) continue;
#pragma unroll
        for (int dx = 0; dx < 2; ++dx) {
            int ix = 2 * ox + dx;
            if (ix >= W) continue;
            m = fmaxf(m, xp[(size_t)iy * W + ix]);
        }
    }
    Y[i] = m;
}

__global__ void maxpool3s1_kernel(const float* __restrict__ X, float* __restrict__ Y,
                                  int BC, int H, int W)
{
    int i = blockIdx.x * blockDim.x + threadIdx.x;
    int tot = BC * H * W;
    if (i >= tot) return;
    int bc = i / (H * W);
    int r  = i - bc * H * W;
    int oy = r / W, ox = r - (r / W) * W;
    const float* xp = X + (size_t)bc * H * W;
    float m = -INFINITY;
#pragma unroll
    for (int dy = -1; dy <= 1; ++dy) {
        int iy = oy + dy;
        if ((unsigned)iy >= (unsigned)H) continue;
#pragma unroll
        for (int dx = -1; dx <= 1; ++dx) {
            int ix = ox + dx;
            if ((unsigned)ix >= (unsigned)W) continue;
            m = fmaxf(m, xp[(size_t)iy * W + ix]);
        }
    }
    Y[i] = m;
}

__global__ void l2norm_kernel(const float* __restrict__ X, const float* __restrict__ g,
                              float* __restrict__ Y, int B, int C, int HW)
{
    int i = blockIdx.x * blockDim.x + threadIdx.x;
    if (i >= B * HW) return;
    int b = i / HW, p = i - b * HW;
    const float* xp = X + (size_t)b * C * HW + p;
    float sum = 0.0f;
    for (int c = 0; c < C; ++c) { float v = xp[(size_t)c * HW]; sum += v * v; }
    float inv = 1.0f / (sqrtf(sum) + 1e-10f);
    float* yp = Y + (size_t)b * C * HW + p;
    for (int c = 0; c < C; ++c) yp[(size_t)c * HW] = g[c] * xp[(size_t)c * HW] * inv;
}

// bilinear (half-pixel centers) upsample of Xs (Hin x Hin) added into Y (Hout x Hout)
__global__ void resize_add_kernel(const float* __restrict__ Xs, float* __restrict__ Y,
                                  int BC, int Hin, int Hout)
{
    int i = blockIdx.x * blockDim.x + threadIdx.x;
    int tot = BC * Hout * Hout;
    if (i >= tot) return;
    int bc = i / (Hout * Hout);
    int r  = i - bc * Hout * Hout;
    int oy = r / Hout, ox = r - (r / Hout) * Hout;
    float scale = (float)Hin / (float)Hout;
    float sy = fminf(fmaxf((oy + 0.5f) * scale - 0.5f, 0.0f), (float)(Hin - 1));
    float sx = fminf(fmaxf((ox + 0.5f) * scale - 0.5f, 0.0f), (float)(Hin - 1));
    int y0 = (int)floorf(sy), x0 = (int)floorf(sx);
    int y1 = min(y0 + 1, Hin - 1), x1 = min(x0 + 1, Hin - 1);
    float wy = sy - (float)y0, wx = sx - (float)x0;
    const float* xp = Xs + (size_t)bc * Hin * Hin;
    float v = (1.0f - wy) * ((1.0f - wx) * xp[y0 * Hin + x0] + wx * xp[y0 * Hin + x1]) +
              wy          * ((1.0f - wx) * xp[y1 * Hin + x0] + wx * xp[y1 * Hin + x1]);
    Y[i] += v;
}

// NCHW -> per-batch NHWC slab written at baseOff inside a row of rowStride floats
__global__ void permute_concat_kernel(const float* __restrict__ X, float* __restrict__ Y,
                                      int B, int C, int H, int W, int rowStride, int baseOff)
{
    int i = blockIdx.x * blockDim.x + threadIdx.x;
    int tot = B * C * H * W;
    if (i >= tot) return;
    int c = i % C;
    int r = i / C;
    int x = r % W; r /= W;
    int y = r % H;
    int b = r / H;
    Y[(size_t)b * rowStride + baseOff + ((size_t)(y * W + x) * C + c)] =
        X[(((size_t)b * C + c) * H + y) * W + x];
}

__global__ void softmax21_kernel(const float* __restrict__ X, float* __restrict__ Y, int total)
{
    int i = blockIdx.x * blockDim.x + threadIdx.x;
    if (i >= total) return;
    const float* xp = X + (size_t)i * NUM_CLASSES;
    float* yp = Y + (size_t)i * NUM_CLASSES;
    float m = xp[0];
#pragma unroll
    for (int c = 1; c < NUM_CLASSES; ++c) m = fmaxf(m, xp[c]);
    float e[NUM_CLASSES], s = 0.0f;
#pragma unroll
    for (int c = 0; c < NUM_CLASSES; ++c) { e[c] = expf(xp[c] - m); s += e[c]; }
    float inv = 1.0f / s;
#pragma unroll
    for (int c = 0; c < NUM_CLASSES; ++c) yp[c] = e[c] * inv;
}

__global__ void decode_kernel(const float* __restrict__ L, const float* __restrict__ db,
                              float* __restrict__ BX, int B, int N)
{
    int i = blockIdx.x * blockDim.x + threadIdx.x;
    if (i >= B * N) return;
    int n = i % N;
    const float* l = L + (size_t)i * 4;
    const float* d = db + (size_t)n * 4;
    float cx = d[0] + l[0] * 0.1f * d[2];
    float cy = d[1] + l[1] * 0.1f * d[3];
    float w  = d[2] * expf(l[2] * 0.2f);
    float h  = d[3] * expf(l[3] * 0.2f);
    float x1 = cx - 0.5f * w, y1 = cy - 0.5f * h;
    float* o = BX + (size_t)i * 4;
    o[0] = x1; o[1] = y1; o[2] = x1 + w; o[3] = y1 + h;
}

__global__ void zero_kernel(float* p, int n)
{
    int i = blockIdx.x * blockDim.x + threadIdx.x;
    if (i < n) p[i] = 0.0f;
}

// One workgroup per (class, batch). Score array lives in LDS (35 KB << 320 KB WGP).
__global__ __launch_bounds__(256)
void nms_kernel(const float* __restrict__ probs, const float* __restrict__ boxes,
                float* __restrict__ out, int B, int N)
{
    __shared__ float s[NPRIOR];
    __shared__ float redv[256];
    __shared__ int   redi[256];
    __shared__ float ts[TOPK];
    __shared__ int   tidx[TOPK];
    __shared__ float bx1[TOPK], by1[TOPK], bx2[TOPK], by2[TOPK], area[TOPK];
    __shared__ int   keep[TOPK];

    const int tid = threadIdx.x;
    const int cls = blockIdx.x + 1;      // classes 1..20
    const int b   = blockIdx.y;

    for (int n = tid; n < N; n += 256) {
        float p = probs[((size_t)b * N + n) * NUM_CLASSES + cls];
        s[n] = (p > 0.01f) ? p : -1.0f;
    }
    __syncthreads();

    // iterative top-200 (value desc, index asc on ties — matches lax.top_k)
    for (int k = 0; k < TOPK; ++k) {
        float best = -1e30f; int bi = N;
        for (int n = tid; n < N; n += 256) {
            float v = s[n];
            if (v > best || (v == best && n < bi)) { best = v; bi = n; }
        }
        redv[tid] = best; redi[tid] = bi;
        __syncthreads();
        for (int off = 128; off > 0; off >>= 1) {
            if (tid < off) {
                float v2 = redv[tid + off]; int i2 = redi[tid + off];
                if (v2 > redv[tid] || (v2 == redv[tid] && i2 < redi[tid])) {
                    redv[tid] = v2; redi[tid] = i2;
                }
            }
            __syncthreads();
        }
        if (tid == 0) {
            int i = redi[0];
            ts[k] = redv[0];
            tidx[k] = (i < N) ? i : 0;
            if (i < N) s[i] = -1e30f;
        }
        __syncthreads();
    }

    for (int k = tid; k < TOPK; k += 256) {
        int i = tidx[k];
        const float* bb = boxes + ((size_t)b * N + i) * 4;
        float x1 = bb[0], y1 = bb[1], x2 = bb[2], y2 = bb[3];
        bx1[k] = x1; by1[k] = y1; bx2[k] = x2; by2[k] = y2;
        area[k] = (x2 - x1) * (y2 - y1);
        keep[k] = (ts[k] > 0.01f) ? 1 : 0;
    }
    __syncthreads();

    for (int i = 0; i < TOPK - 1; ++i) {
        if (keep[i]) {
            for (int j = i + 1 + tid; j < TOPK; j += 256) {
                float iw = fmaxf(fminf(bx2[i], bx2[j]) - fmaxf(bx1[i], bx1[j]), 0.0f);
                float ih = fmaxf(fminf(by2[i], by2[j]) - fmaxf(by1[i], by1[j]), 0.0f);
                float inter = iw * ih;
                float iou = inter / fmaxf(area[i] + area[j] - inter, 1e-12f);
                if (iou > 0.45f) keep[j] = 0;
            }
        }
        __syncthreads();
    }

    if (tid == 0) {
        int r = 0;
        for (int k = 0; k < TOPK; ++k) {
            if (keep[k]) {
                float* dst = out + (((size_t)b * NUM_CLASSES + cls) * TOPK + r) * 5;
                dst[0] = ts[k]; dst[1] = bx1[k]; dst[2] = by1[k];
                dst[3] = bx2[k]; dst[4] = by2[k];
                ++r;
            }
        }
    }
}

// ---------------------------------------------------------------------------
// Host-side graph
// ---------------------------------------------------------------------------
static inline int conv_out_sz(int H, int k, int s, int p, int d)
{
    return (H + 2 * p - d * (k - 1) - 1) / s + 1;
}

static void launch_conv(hipStream_t st, const float* X, void* const* din, int wi,
                        float* Y, int B, int Cin, int H, int Cout,
                        int k, int s, int p, int d, int relu, int* OHo)
{
    int OH = conv_out_sz(H, k, s, p, d);
    dim3 g((unsigned)((B * OH * OH + 31) / 32), (unsigned)((Cout + 15) / 16));
    conv_wmma<<<g, 32, 0, st>>>(X, (const float*)din[wi], (const float*)din[wi + 1], Y,
                                B, Cin, H, H, Cout, OH, OH, k, k, s, p, d, relu);
    if (OHo) *OHo = OH;
}

static void launch_pool2(hipStream_t st, const float* X, float* Y, int BC, int H,
                         bool ceilm, int* OHo)
{
    int oh = ceilm ? (H + 1) / 2 : H / 2;
    int n = BC * oh * oh;
    maxpool2_kernel<<<(n + 255) / 256, 256, 0, st>>>(X, Y, BC, H, H, oh, oh);
    *OHo = oh;
}

extern "C" void kernel_launch(void* const* d_in, const int* in_sizes, int n_in,
                              void* d_out, int out_size, void* d_ws, size_t ws_size,
                              hipStream_t stream)
{
    (void)in_sizes; (void)n_in; (void)ws_size;
    const int B = 16;
    char* wsp = (char*)d_ws;
    size_t used = 0;
    auto alloc = [&](size_t nfloats) -> float* {
        float* p = (float*)(wsp + used);
        used += ((nfloats * sizeof(float)) + 255) & ~(size_t)255;
        return p;
    };

    float* bufA = alloc((size_t)B * 64 * 300 * 300);
    float* bufB = alloc((size_t)B * 64 * 300 * 300);
    float* S0 = alloc((size_t)B * 512 * 38 * 38);
    float* S1 = alloc((size_t)B * 1024 * 19 * 19);
    float* S2 = alloc((size_t)B * 512 * 10 * 10);
    float* S3 = alloc((size_t)B * 256 * 5 * 5);
    float* S4 = alloc((size_t)B * 256 * 3 * 3);
    float* S5 = alloc((size_t)B * 256 * 1 * 1);
    float* F0 = alloc((size_t)B * 256 * 38 * 38);
    float* F1 = alloc((size_t)B * 256 * 19 * 19);
    float* F2 = alloc((size_t)B * 256 * 10 * 10);
    float* F3 = alloc((size_t)B * 256 * 5 * 5);
    float* F4 = alloc((size_t)B * 256 * 3 * 3);
    float* F5 = alloc((size_t)B * 256 * 1 * 1);
    float* LOC  = alloc((size_t)B * NPRIOR * 4);
    float* CONF = alloc((size_t)B * NPRIOR * NUM_CLASSES);
    float* PROB = alloc((size_t)B * NPRIOR * NUM_CLASSES);
    float* BOX  = alloc((size_t)B * NPRIOR * 4);

    const float* x = (const float*)d_in[0];
    int H;

    // ---- VGG backbone ----
    launch_conv(stream, x,    d_in, 1,  bufA, B, 3,   300, 64,  3, 1, 1, 1, 1, &H);
    launch_conv(stream, bufA, d_in, 3,  bufB, B, 64,  300, 64,  3, 1, 1, 1, 1, &H);
    launch_pool2(stream, bufB, bufA, B * 64, 300, false, &H);              // 150
    launch_conv(stream, bufA, d_in, 5,  bufB, B, 64,  150, 128, 3, 1, 1, 1, 1, &H);
    launch_conv(stream, bufB, d_in, 7,  bufA, B, 128, 150, 128, 3, 1, 1, 1, 1, &H);
    launch_pool2(stream, bufA, bufB, B * 128, 150, false, &H);             // 75
    launch_conv(stream, bufB, d_in, 9,  bufA, B, 128, 75,  256, 3, 1, 1, 1, 1, &H);
    launch_conv(stream, bufA, d_in, 11, bufB, B, 256, 75,  256, 3, 1, 1, 1, 1, &H);
    launch_conv(stream, bufB, d_in, 13, bufA, B, 256, 75,  256, 3, 1, 1, 1, 1, &H);
    launch_pool2(stream, bufA, bufB, B * 256, 75, true, &H);               // 38 (ceil)
    launch_conv(stream, bufB, d_in, 15, bufA, B, 256, 38,  512, 3, 1, 1, 1, 1, &H);
    launch_conv(stream, bufA, d_in, 17, bufB, B, 512, 38,  512, 3, 1, 1, 1, 1, &H);
    launch_conv(stream, bufB, d_in, 19, bufA, B, 512, 38,  512, 3, 1, 1, 1, 1, &H);
    {   // L2Norm branch -> S0 (pre-FPN source 0)
        int n = B * 38 * 38;
        l2norm_kernel<<<(n + 255) / 256, 256, 0, stream>>>(bufA, (const float*)d_in[83],
                                                           S0, B, 512, 38 * 38);
    }
    launch_pool2(stream, bufA, bufB, B * 512, 38, false, &H);              // 19
    launch_conv(stream, bufB, d_in, 21, bufA, B, 512, 19,  512, 3, 1, 1, 1, 1, &H);
    launch_conv(stream, bufA, d_in, 23, bufB, B, 512, 19,  512, 3, 1, 1, 1, 1, &H);
    launch_conv(stream, bufB, d_in, 25, bufA, B, 512, 19,  512, 3, 1, 1, 1, 1, &H);
    {   // maxpool 3x3 stride1 pad1
        int n = B * 512 * 19 * 19;
        maxpool3s1_kernel<<<(n + 255) / 256, 256, 0, stream>>>(bufA, bufB, B * 512, 19, 19);
    }
    launch_conv(stream, bufB, d_in, 27, bufA, B, 512,  19, 1024, 3, 1, 6, 6, 1, &H); // dil6
    launch_conv(stream, bufA, d_in, 29, S1,   B, 1024, 19, 1024, 1, 1, 0, 1, 1, &H);

    // ---- Extras ----
    launch_conv(stream, S1,   d_in, 31, bufA, B, 1024, 19, 256, 1, 1, 0, 1, 1, &H);
    launch_conv(stream, bufA, d_in, 33, S2,   B, 256,  19, 512, 3, 2, 1, 1, 1, &H);  // 10
    launch_conv(stream, S2,   d_in, 35, bufA, B, 512,  10, 128, 1, 1, 0, 1, 1, &H);
    launch_conv(stream, bufA, d_in, 37, S3,   B, 128,  10, 256, 3, 2, 1, 1, 1, &H);  // 5
    launch_conv(stream, S3,   d_in, 39, bufA, B, 256,  5,  128, 1, 1, 0, 1, 1, &H);
    launch_conv(stream, bufA, d_in, 41, S4,   B, 128,  5,  256, 3, 1, 0, 1, 1, &H);  // 3
    launch_conv(stream, S4,   d_in, 43, bufA, B, 256,  3,  128, 1, 1, 0, 1, 1, &H);
    launch_conv(stream, bufA, d_in, 45, S5,   B, 128,  3,  256, 3, 1, 0, 1, 1, &H);  // 1

    // ---- FPN (1x1 convs, no relu) + top-down bilinear adds ----
    launch_conv(stream, S5, d_in, 47, F5, B, 256,  1,  256, 1, 1, 0, 1, 0, &H);
    launch_conv(stream, S4, d_in, 49, F4, B, 256,  3,  256, 1, 1, 0, 1, 0, &H);
    { int n = B * 256 * 3 * 3;
      resize_add_kernel<<<(n + 255) / 256, 256, 0, stream>>>(F5, F4, B * 256, 1, 3); }
    launch_conv(stream, S3, d_in, 51, F3, B, 256,  5,  256, 1, 1, 0, 1, 0, &H);
    { int n = B * 256 * 5 * 5;
      resize_add_kernel<<<(n + 255) / 256, 256, 0, stream>>>(F4, F3, B * 256, 3, 5); }
    launch_conv(stream, S2, d_in, 53, F2, B, 512,  10, 256, 1, 1, 0, 1, 0, &H);
    { int n = B * 256 * 10 * 10;
      resize_add_kernel<<<(n + 255) / 256, 256, 0, stream>>>(F3, F2, B * 256, 5, 10); }
    launch_conv(stream, S1, d_in, 55, F1, B, 1024, 19, 256, 1, 1, 0, 1, 0, &H);
    { int n = B * 256 * 19 * 19;
      resize_add_kernel<<<(n + 255) / 256, 256, 0, stream>>>(F2, F1, B * 256, 10, 19); }
    launch_conv(stream, S0, d_in, 57, F0, B, 512,  38, 256, 1, 1, 0, 1, 0, &H);
    { int n = B * 256 * 38 * 38;
      resize_add_kernel<<<(n + 255) / 256, 256, 0, stream>>>(F1, F0, B * 256, 19, 38); }

    // ---- loc / conf heads + NHWC concat ----
    const float* Fs[6] = {F0, F1, F2, F3, F4, F5};
    const int hs[6]   = {38, 19, 10, 5, 3, 1};
    const int nb[6]   = {4, 6, 6, 6, 4, 4};
    const int poff[6] = {0, 5776, 7942, 8542, 8692, 8728};
    for (int si = 0; si < 6; ++si) {
        int C4 = nb[si] * 4, C21 = nb[si] * NUM_CLASSES;
        launch_conv(stream, Fs[si], d_in, 59 + 2 * si, bufA, B, 256, hs[si], C4,
                    3, 1, 1, 1, 0, &H);
        int n = B * hs[si] * hs[si] * C4;
        permute_concat_kernel<<<(n + 255) / 256, 256, 0, stream>>>(
            bufA, LOC, B, C4, hs[si], hs[si], NPRIOR * 4, poff[si] * 4);
        launch_conv(stream, Fs[si], d_in, 71 + 2 * si, bufA, B, 256, hs[si], C21,
                    3, 1, 1, 1, 0, &H);
        n = B * hs[si] * hs[si] * C21;
        permute_concat_kernel<<<(n + 255) / 256, 256, 0, stream>>>(
            bufA, CONF, B, C21, hs[si], hs[si], NPRIOR * NUM_CLASSES,
            poff[si] * NUM_CLASSES);
    }

    // ---- detection ----
    {
        int n = B * NPRIOR;
        softmax21_kernel<<<(n + 255) / 256, 256, 0, stream>>>(CONF, PROB, n);
        decode_kernel<<<(n + 255) / 256, 256, 0, stream>>>(LOC, (const float*)d_in[84],
                                                           BOX, B, NPRIOR);
        zero_kernel<<<(out_size + 255) / 256, 256, 0, stream>>>((float*)d_out, out_size);
        dim3 gn(NUM_CLASSES - 1, B);
        nms_kernel<<<gn, 256, 0, stream>>>(PROB, BOX, (float*)d_out, B, NPRIOR);
    }
}
